// VDPWIModel_22368189678452
// MI455X (gfx1250) — matrix operand order repaired and checked by reading, not compiled, not run on hardware
//
#include <hip/hip_runtime.h>
#include <hip/hip_bf16.h>
#include <math.h>

// ---------------------------------------------------------------------------
// Problem constants
// ---------------------------------------------------------------------------
constexpr int Bz   = 128;   // batch
constexpr int Sq   = 32;    // sequence length
constexpr int DIMv = 300;   // input feature dim
constexpr int Hh   = 256;   // LSTM hidden
constexpr int G4   = 1024;  // 4*H gate width
constexpr int K1P  = 320;   // pad(300) to 32
constexpr int K2P  = 832;   // pad(812)
constexpr int K3P  = 1344;  // pad(1324)
constexpr int M1   = 4 * Bz * Sq;  // 16384 (x1,x2,flip(x1),flip(x2))
constexpr int M2   = 2 * Bz * Sq;  // 8192  (sent1, sent2)

typedef __attribute__((ext_vector_type(16))) _Float16 v16h;
typedef __attribute__((ext_vector_type(8)))  _Float16 v8h;
typedef __attribute__((ext_vector_type(8)))  float    v8f;

// ---------------------------------------------------------------------------
// Workspace layout (byte offsets, 256-aligned, phase-reused)
// ---------------------------------------------------------------------------
constexpr size_t AL(size_t x) { return (x + 255) & ~(size_t)255; }

constexpr size_t O_W1IH = 0;
constexpr size_t O_W1HH  = AL(O_W1IH  + (size_t)1024*320*2);
constexpr size_t O_W2FIH = AL(O_W1HH  + (size_t)1024*256*2);
constexpr size_t O_W2BIH = AL(O_W2FIH + (size_t)1024*832*2);
constexpr size_t O_W2FHH = AL(O_W2BIH + (size_t)1024*832*2);
constexpr size_t O_W2BHH = AL(O_W2FHH + (size_t)1024*256*2);
constexpr size_t O_W3FIH = AL(O_W2BHH + (size_t)1024*256*2);
constexpr size_t O_W3BIH = AL(O_W3FIH + (size_t)1024*1344*2);
constexpr size_t O_W3FHH = AL(O_W3BIH + (size_t)1024*1344*2);
constexpr size_t O_W3BHH = AL(O_W3FHH + (size_t)1024*256*2);
constexpr size_t O_CW1   = AL(O_W3BHH + (size_t)1024*256*2);   // 128 x 128
constexpr size_t O_CW2   = AL(O_CW1 + (size_t)128*128*2);      // 192 x 1152
constexpr size_t O_CW3   = AL(O_CW2 + (size_t)192*1152*2);     // 192 x 1504
constexpr size_t O_CW4   = AL(O_CW3 + (size_t)192*1504*2);     // 192 x 1728
constexpr size_t O_CW5   = AL(O_CW4 + (size_t)192*1728*2);     // 128 x 1728
constexpr size_t O_RW1   = AL(O_CW5 + (size_t)128*1728*2);     // 192 x 128
constexpr size_t O_RW2   = AL(O_RW1 + (size_t)192*128*2);      // 192 x 1504
constexpr size_t O_DW    = AL(O_RW2 + (size_t)192*1504*2);     // 128 x 128
constexpr size_t O_H1    = AL(O_DW  + (size_t)128*128*2);      // (512,32,256) f16
constexpr size_t O_H2F   = AL(O_H1  + (size_t)16384*256*2);
constexpr size_t O_H2B   = AL(O_H2F + (size_t)8192*256*2);
constexpr size_t O_H3F   = AL(O_H2B + (size_t)8192*256*2);
constexpr size_t O_H3B   = AL(O_H3F + (size_t)8192*256*2);
constexpr size_t O_GT    = AL(O_H3B + (size_t)8192*256*2);     // Gtmp 512x1024 f32
constexpr size_t O_CS    = AL(O_GT  + (size_t)512*1024*4);     // c-state 512x256 f32
constexpr size_t O_SEQ1  = AL(O_CS  + (size_t)512*256*4);      // (B,32,512) f16
constexpr size_t O_SEQ2  = AL(O_SEQ1+ (size_t)128*32*512*2);
constexpr size_t O_DOTS  = AL(O_SEQ2+ (size_t)128*32*512*2);   // 4x(B,32,32) f32
constexpr size_t O_ST    = AL(O_DOTS+ (size_t)4*128*32*32*4);  // stats 2xBx32x3 f32
constexpr size_t O_SIM   = AL(O_ST  + (size_t)2*128*32*3*4);   // (B,12,32,32) f32
constexpr size_t O_SEL   = AL(O_SIM + (size_t)128*12*32*32*4); // (B,2,32,32) f32
constexpr size_t O_FOC   = AL(O_SEL + (size_t)128*2*32*32*4);  // (B,12,32,32) f32
constexpr size_t O_A1    = AL(O_FOC + (size_t)128*12*32*32*4); // 10.5MB region
constexpr size_t O_GIN   = AL(O_A1  + (size_t)16384*320*2);    // 64MB region
constexpr size_t O_RA    = AL(O_GIN + (size_t)16384*1024*4);   // 44MB region
constexpr size_t RA_HALF = (size_t)8192*1344*2;
// phase-reused sub-offsets
constexpr size_t O_GIN2F = O_GIN;
constexpr size_t O_GIN2B = O_GIN + (size_t)8192*1024*4;
constexpr size_t O_A2F = O_RA, O_A2B = O_RA + RA_HALF;
constexpr size_t O_A3F = O_RA, O_A3B = O_RA + RA_HALF;
constexpr size_t O_C1O = O_GIN;                // (B,128,32,32) f32 = 64MB exactly
constexpr size_t O_P1  = O_RA;                 // (B,128,16,16)
constexpr size_t O_R1O = O_RA + RA_HALF;       // (B,164,16,16)
constexpr size_t O_C2O = O_GIN;                // (B,164,16,16)
constexpr size_t O_P2  = O_A1;                 // (B,164,8,8)
constexpr size_t O_C3O = O_RA;                 // (B,192,8,8)
constexpr size_t O_P3  = O_RA + RA_HALF;       // (B,192,4,4)
constexpr size_t O_R2O = O_GIN;                // (B,192,4,4)
constexpr size_t O_C4O = O_GIN + (size_t)4*1024*1024;
constexpr size_t O_P4  = O_RA;                 // (B,192,2,2)
constexpr size_t O_C5O = O_GIN;                // (B,128,2,2)
constexpr size_t O_P5  = O_RA + (size_t)1024*1024;   // (B,128) f32
constexpr size_t O_FH  = O_RA + (size_t)2*1024*1024; // (B,128) f16
constexpr size_t O_FC  = O_RA + (size_t)3*1024*1024; // (B,128) f32

// ---------------------------------------------------------------------------
// WMMA fragment loaders (wave32 layouts per CDNA5 ISA 7.12.2)
// A 16x32 f16: lanes 0-15 -> M=lane; elems 0..7 = K[8*hi .. 8*hi+8),
//              elems 8..15 = K[16+8*hi ..), hi = lane>>4.
// B 32x16 f16: lane -> N=lane&15; elems 0..7 = K[8*hi ..), elems 8..15 = K[16+8*hi ..).
// C/D 16x16 f32: lane -> N=lane&15; elem v -> M = v + 8*(lane>>4).
// ---------------------------------------------------------------------------
__device__ __forceinline__ v16h load_a_f16(const _Float16* base, int rowStride,
                                           int m, int lane, int k0) {
  const _Float16* p = base + (size_t)(m + (lane & 15)) * rowStride + k0 + ((lane >> 4) << 3);
  v8h lo = *(const v8h*)p;
  v8h hi = *(const v8h*)(p + 16);
  v16h r;
#pragma unroll
  for (int i = 0; i < 8; ++i) { r[i] = lo[i]; r[i + 8] = hi[i]; }
  return r;
}

__device__ __forceinline__ v16h load_b_f16(const _Float16* base, int rowStride,
                                           int n, int lane, int k0) {
  const _Float16* p = base + (size_t)(n + (lane & 15)) * rowStride + k0 + ((lane >> 4) << 3);
  v8h lo = *(const v8h*)p;
  v8h hi = *(const v8h*)(p + 16);
  v16h r;
#pragma unroll
  for (int i = 0; i < 8; ++i) { r[i] = lo[i]; r[i + 8] = hi[i]; }
  return r;
}

#define WMMA_F16(a, b, c) \
  __builtin_amdgcn_wmma_f32_16x16x32_f16(false, (a), false, (b), (short)0, (c), false, false)

__device__ __forceinline__ float sigm(float x) { return 1.f / (1.f + __expf(-x)); }

// ---------------------------------------------------------------------------
// Generic GEMM: C[M,N] = A[M,Kp](f16) x W[N,Kp]^T(f16) (+bias1+bias2) (+relu)
// 16x64 tile per wave (4 accumulators share one A fragment), K-loop
// software-pipelined one stage so global loads overlap the WMMA group.
// Requires N % 64 == 0 and Kp % 32 == 0.
// ---------------------------------------------------------------------------
__global__ void gemm_f16(const _Float16* __restrict__ A, const _Float16* __restrict__ W,
                         float* __restrict__ C, const float* __restrict__ bias1,
                         const float* __restrict__ bias2, int M, int N, int Kp, int relu) {
  int lane = threadIdx.x & 31;
  int wt = blockIdx.x * 4 + (threadIdx.x >> 5);
  int ng = N >> 6;
  int total = (M >> 4) * ng;
  if (wt >= total) return;                 // wave-uniform exit, EXEC stays full
  int tm = (wt / ng) << 4;
  int tn = (wt % ng) << 6;
  v8f a0 = {}, a1 = {}, a2 = {}, a3 = {};
  // pipeline stage 0
  v16h a  = load_a_f16(A, Kp, tm, lane, 0);
  v16h b0 = load_b_f16(W, Kp, tn,      lane, 0);
  v16h b1 = load_b_f16(W, Kp, tn + 16, lane, 0);
  v16h b2 = load_b_f16(W, Kp, tn + 32, lane, 0);
  v16h b3 = load_b_f16(W, Kp, tn + 48, lane, 0);
  int k0 = 0;
  for (; k0 + 32 < Kp; k0 += 32) {
    int kn = k0 + 32;
    __builtin_prefetch(A + (size_t)(tm + (lane & 15)) * Kp + kn + 32, 0, 1);
    v16h an  = load_a_f16(A, Kp, tm, lane, kn);
    v16h bn0 = load_b_f16(W, Kp, tn,      lane, kn);
    v16h bn1 = load_b_f16(W, Kp, tn + 16, lane, kn);
    v16h bn2 = load_b_f16(W, Kp, tn + 32, lane, kn);
    v16h bn3 = load_b_f16(W, Kp, tn + 48, lane, kn);
    a0 = WMMA_F16(a, b0, a0);
    a1 = WMMA_F16(a, b1, a1);
    a2 = WMMA_F16(a, b2, a2);
    a3 = WMMA_F16(a, b3, a3);
    a = an; b0 = bn0; b1 = bn1; b2 = bn2; b3 = bn3;
  }
  a0 = WMMA_F16(a, b0, a0);
  a1 = WMMA_F16(a, b1, a1);
  a2 = WMMA_F16(a, b2, a2);
  a3 = WMMA_F16(a, b3, a3);
#pragma unroll
  for (int j = 0; j < 4; ++j) {
    v8f acc = (j == 0) ? a0 : (j == 1) ? a1 : (j == 2) ? a2 : a3;
    int col = tn + j * 16 + (lane & 15);
    float bb = (bias1 ? bias1[col] : 0.f) + (bias2 ? bias2[col] : 0.f);
#pragma unroll
    for (int v = 0; v < 8; ++v) {
      int row = tm + v + ((lane >> 4) << 3);
      float x = acc[v] + bb;
      if (relu) x = fmaxf(x, 0.f);
      C[(size_t)row * N + col] = x;
    }
  }
}

// ---------------------------------------------------------------------------
// LSTM recurrent step: Gtmp[batch,1024] = Gin[:,t,:] + h_{t-1} @ Whh^T
// h history kept in f16 (batch, Sq, 256); 16x64 tile per wave; pipelined.
// ---------------------------------------------------------------------------
__global__ void lstm_step_gemm(const float* __restrict__ Gin, const _Float16* __restrict__ Hbuf,
                               const _Float16* __restrict__ Whh, float* __restrict__ Gtmp,
                               int batch, int t) {
  int lane = threadIdx.x & 31;
  int wt = blockIdx.x * 4 + (threadIdx.x >> 5);
  const int ng = G4 >> 6;  // 16
  int total = (batch >> 4) * ng;
  if (wt >= total) return;
  int tm = (wt / ng) << 4;
  int tn = (wt % ng) << 6;
  v8f a0 = {}, a1 = {}, a2 = {}, a3 = {};
  if (t > 0) {
    const _Float16* hb = Hbuf + (size_t)(t - 1) * Hh;  // row m -> m*Sq*Hh
    v16h a  = load_a_f16(hb, Sq * Hh, tm, lane, 0);
    v16h b0 = load_b_f16(Whh, Hh, tn,      lane, 0);
    v16h b1 = load_b_f16(Whh, Hh, tn + 16, lane, 0);
    v16h b2 = load_b_f16(Whh, Hh, tn + 32, lane, 0);
    v16h b3 = load_b_f16(Whh, Hh, tn + 48, lane, 0);
    int k0 = 0;
#pragma unroll 1
    for (; k0 + 32 < Hh; k0 += 32) {
      int kn = k0 + 32;
      v16h an  = load_a_f16(hb, Sq * Hh, tm, lane, kn);
      v16h bn0 = load_b_f16(Whh, Hh, tn,      lane, kn);
      v16h bn1 = load_b_f16(Whh, Hh, tn + 16, lane, kn);
      v16h bn2 = load_b_f16(Whh, Hh, tn + 32, lane, kn);
      v16h bn3 = load_b_f16(Whh, Hh, tn + 48, lane, kn);
      a0 = WMMA_F16(a, b0, a0);
      a1 = WMMA_F16(a, b1, a1);
      a2 = WMMA_F16(a, b2, a2);
      a3 = WMMA_F16(a, b3, a3);
      a = an; b0 = bn0; b1 = bn1; b2 = bn2; b3 = bn3;
    }
    a0 = WMMA_F16(a, b0, a0);
    a1 = WMMA_F16(a, b1, a1);
    a2 = WMMA_F16(a, b2, a2);
    a3 = WMMA_F16(a, b3, a3);
  }
#pragma unroll
  for (int j = 0; j < 4; ++j) {
    v8f acc = (j == 0) ? a0 : (j == 1) ? a1 : (j == 2) ? a2 : a3;
    int col = tn + j * 16 + (lane & 15);
#pragma unroll
    for (int v = 0; v < 8; ++v) {
      int row = tm + v + ((lane >> 4) << 3);
      Gtmp[(size_t)row * G4 + col] = acc[v] + Gin[((size_t)row * Sq + t) * G4 + col];
    }
  }
}

// Fused LSTM gates + state update; writes h directly as f16.
__global__ void lstm_pointwise(const float* __restrict__ Gtmp, float* __restrict__ Cst,
                               _Float16* __restrict__ Hbuf, int batch, int t) {
  int idx = blockIdx.x * blockDim.x + threadIdx.x;
  if (idx >= batch * Hh) return;
  int m = idx / Hh, j = idx - m * Hh;
  const float* g = Gtmp + (size_t)m * G4;
  float gi = g[j], gf = g[j + Hh], gg = g[j + 2 * Hh], go = g[j + 3 * Hh];
  float c = t ? Cst[idx] : 0.f;
  float cn = sigm(gf) * c + sigm(gi) * tanhf(gg);
  float h = sigm(go) * tanhf(cn);
  Cst[idx] = cn;
  Hbuf[((size_t)m * Sq + t) * Hh + j] = (_Float16)h;
}

// ---------------------------------------------------------------------------
// Activation assembly kernels (f32 -> padded f16 GEMM operands)
// ---------------------------------------------------------------------------
__global__ void asm_a1_k(const float* __restrict__ s1, const float* __restrict__ s2,
                         _Float16* __restrict__ A) {
  int idx = blockIdx.x * blockDim.x + threadIdx.x;
  if (idx >= M1 * K1P) return;
  int m = idx / K1P, k = idx - (idx / K1P) * K1P;
  int q = m / (Bz * Sq);
  int r = m - q * (Bz * Sq);
  int b = r / Sq, s = r - (r / Sq) * Sq;
  float v = 0.f;
  if (k < DIMv) {
    int ss = (q >= 2) ? (Sq - 1 - s) : s;
    const float* src = (q & 1) ? s2 : s1;
    v = src[((size_t)b * DIMv + k) * Sq + ss];
  }
  A[idx] = (_Float16)v;
}

__global__ void asm_a2_k(const float* __restrict__ s1, const float* __restrict__ s2,
                         const _Float16* __restrict__ H1, _Float16* __restrict__ A, int flip) {
  long idx = (long)blockIdx.x * blockDim.x + threadIdx.x;
  if (idx >= (long)M2 * K2P) return;
  int m = (int)(idx / K2P), k = (int)(idx - (long)m * K2P);
  int p = m / (Bz * Sq);
  int r = m - p * (Bz * Sq);
  int b = r / Sq, s0 = r - (r / Sq) * Sq;
  int s = flip ? (Sq - 1 - s0) : s0;
  _Float16 v = (_Float16)0.f;
  if (k < DIMv) {
    const float* src = p ? s2 : s1;
    v = (_Float16)src[((size_t)b * DIMv + k) * Sq + s];
  } else if (k < DIMv + Hh) {
    v = H1[(((size_t)(p * Bz + b)) * Sq + s) * Hh + (k - DIMv)];
  } else if (k < DIMv + 2 * Hh) {
    v = H1[(((size_t)((2 + p) * Bz + b)) * Sq + (Sq - 1 - s)) * Hh + (k - DIMv - Hh)];
  }
  A[idx] = v;
}

__global__ void asm_a3_k(const float* __restrict__ s1, const float* __restrict__ s2,
                         const _Float16* __restrict__ H1, const _Float16* __restrict__ H2f,
                         const _Float16* __restrict__ H2b, _Float16* __restrict__ A, int flip) {
  long idx = (long)blockIdx.x * blockDim.x + threadIdx.x;
  if (idx >= (long)M2 * K3P) return;
  int m = (int)(idx / K3P), k = (int)(idx - (long)m * K3P);
  int p = m / (Bz * Sq);
  int r = m - p * (Bz * Sq);
  int b = r / Sq, s0 = r - (r / Sq) * Sq;
  int s = flip ? (Sq - 1 - s0) : s0;
  _Float16 v = (_Float16)0.f;
  if (k < DIMv) {
    const float* src = p ? s2 : s1;
    v = (_Float16)src[((size_t)b * DIMv + k) * Sq + s];
  } else if (k < DIMv + Hh) {
    v = H1[(((size_t)(p * Bz + b)) * Sq + s) * Hh + (k - DIMv)];
  } else if (k < DIMv + 2 * Hh) {
    v = H1[(((size_t)((2 + p) * Bz + b)) * Sq + (Sq - 1 - s)) * Hh + (k - DIMv - Hh)];
  } else if (k < DIMv + 3 * Hh) {
    v = H2f[(((size_t)(p * Bz + b)) * Sq + s) * Hh + (k - DIMv - 2 * Hh)];
  } else if (k < DIMv + 4 * Hh) {
    v = H2b[(((size_t)(p * Bz + b)) * Sq + (Sq - 1 - s)) * Hh + (k - DIMv - 3 * Hh)];
  }
  A[idx] = v;
}

__global__ void build_seq_k(const _Float16* __restrict__ Hf, const _Float16* __restrict__ Hb,
                            _Float16* __restrict__ seq, int p) {
  int idx = blockIdx.x * blockDim.x + threadIdx.x;
  if (idx >= Bz * Sq * 512) return;
  int d = idx & 511;
  int t = idx >> 9;
  int s = t % Sq, b = t / Sq;
  _Float16 v;
  if (d < Hh) v = Hf[(((size_t)(p * Bz + b)) * Sq + s) * Hh + d];
  else        v = Hb[(((size_t)(p * Bz + b)) * Sq + (Sq - 1 - s)) * Hh + (d - Hh)];
  seq[idx] = v;
}

// ---------------------------------------------------------------------------
// Similarity cube: batched WMMA dot-products (ff/fb/bf/bb over the 256-d halves)
// ---------------------------------------------------------------------------
__global__ void dots_wmma_k(const _Float16* __restrict__ s1, const _Float16* __restrict__ s2,
                            float* __restrict__ dots) {
  int lane = threadIdx.x & 31;
  int tile = blockIdx.x * 4 + (threadIdx.x >> 5);
  int total = Bz * 4 * 4;  // 4 dot types x (2x2 tiles of 32x32)
  if (tile >= total) return;
  int b = tile >> 4;
  int rem = tile & 15;
  int typ = rem >> 2;   // 0:ff 1:fb 2:bf 3:bb
  int tt = rem & 3;
  int tm = (tt >> 1) << 4, tn = (tt & 1) << 4;
  const _Float16* abase = s1 + (size_t)b * 32 * 512 + ((typ >> 1) ? Hh : 0);
  const _Float16* bbase = s2 + (size_t)b * 32 * 512 + ((typ & 1) ? Hh : 0);
  v8f acc = {};
  for (int k0 = 0; k0 < Hh; k0 += 32) {
    v16h a = load_a_f16(abase, 512, tm, lane, k0);
    v16h bf = load_b_f16(bbase, 512, tn, lane, k0);
    acc = WMMA_F16(a, bf, acc);
  }
  int col = tn + (lane & 15);
#pragma unroll
  for (int v = 0; v < 8; ++v) {
    int row = tm + v + ((lane >> 4) << 3);
    dots[(((size_t)typ * Bz + b) * 32 + row) * 32 + col] = acc[v];
  }
}

// per-row norms^2 of f/b halves and the f.b self-dot (for the sum-variant norm)
__global__ void rowstats_k(const _Float16* __restrict__ s1, const _Float16* __restrict__ s2,
                           float* __restrict__ st) {
  int idx = blockIdx.x * blockDim.x + threadIdx.x;
  if (idx >= 2 * Bz * 32) return;
  int sq = idx / (Bz * 32);
  int r = idx - sq * (Bz * 32);
  const _Float16* p = (sq ? s2 : s1) + (size_t)r * 512;
  float nf = 0.f, nb = 0.f, sf = 0.f;
  for (int d = 0; d < Hh; ++d) {
    float a = (float)p[d], c = (float)p[d + Hh];
    nf += a * a; nb += c * c; sf += a * c;
  }
  st[idx * 3 + 0] = nf; st[idx * 3 + 1] = nb; st[idx * 3 + 2] = sf;
}

__device__ __forceinline__ void write_group(float* o, int g, float d, float na2, float nb2) {
  float na = sqrtf(fmaxf(na2, 0.f)), nb = sqrtf(fmaxf(nb2, 0.f));
  o[(size_t)(g * 3 + 0) * 1024] = d;
  o[(size_t)(g * 3 + 1) * 1024] = d / (na * nb + 1e-8f);
  o[(size_t)(g * 3 + 2) * 1024] = sqrtf(fmaxf(na2 + nb2 - 2.f * d, 0.f));  // ||a-b||
}

__global__ void sim_assemble_k(const float* __restrict__ dots, const float* __restrict__ st,
                               float* __restrict__ sim) {
  int idx = blockIdx.x * blockDim.x + threadIdx.x;
  if (idx >= Bz * 1024) return;
  int b = idx >> 10;
  int r = idx & 1023;
  int i = r >> 5, j = r & 31;
  float ff = dots[(((size_t)0 * Bz + b) * 32 + i) * 32 + j];
  float fb = dots[(((size_t)1 * Bz + b) * 32 + i) * 32 + j];
  float bf = dots[(((size_t)2 * Bz + b) * 32 + i) * 32 + j];
  float bb = dots[(((size_t)3 * Bz + b) * 32 + i) * 32 + j];
  const float* t1 = st + (((size_t)0 * Bz + b) * 32 + i) * 3;
  const float* t2 = st + (((size_t)1 * Bz + b) * 32 + j) * 3;
  float n1f = t1[0], n1b = t1[1], x1 = t1[2];
  float n2f = t2[0], n2b = t2[1], x2 = t2[2];
  float* o = sim + (size_t)b * 12 * 1024 + i * 32 + j;
  write_group(o, 0, ff + bb, n1f + n1b, n2f + n2b);                       // full 512-d
  write_group(o, 1, ff, n1f, n2f);                                        // fwd halves
  write_group(o, 2, bb, n1b, n2b);                                        // bwd halves
  write_group(o, 3, ff + fb + bf + bb, n1f + n1b + 2.f * x1, n2f + n2b + 2.f * x2);  // f+b
}

// ---------------------------------------------------------------------------
// Greedy matching mask: one wave per (batch, channel in {9,10}); 32 sequential
// argmax rounds over a 32x32 score matrix held in LDS.
// ---------------------------------------------------------------------------
__global__ void greedy_k(const float* __restrict__ sim, float* __restrict__ sel) {
  __shared__ float mm[1024];
  __shared__ float bv[32];
  __shared__ int bi[32];
  __shared__ int rc[3];
  int b = blockIdx.x >> 1, which = blockIdx.x & 1;
  int ch = 9 + which;
  int lane = threadIdx.x;
  const float* src = sim + ((size_t)b * 12 + ch) * 1024;
  float* selp = sel + ((size_t)b * 2 + which) * 1024;
  for (int e = 0; e < 32; ++e) { mm[lane * 32 + e] = src[lane * 32 + e]; selp[lane * 32 + e] = 0.f; }
  __syncthreads();
  for (int it = 0; it < 32; ++it) {
    float best = -1e30f; int bidx = 0;
    for (int e = 0; e < 32; ++e) {
      float v = mm[lane * 32 + e];
      if (v > best) { best = v; bidx = lane * 32 + e; }   // strict > keeps first max
    }
    bv[lane] = best; bi[lane] = bidx;
    __syncthreads();
    if (lane == 0) {
      float B0 = bv[0]; int I0 = bi[0];
      for (int l = 1; l < 32; ++l) if (bv[l] > B0) { B0 = bv[l]; I0 = bi[l]; }
      rc[0] = I0 >> 5; rc[1] = I0 & 31; rc[2] = (B0 >= -5000.f) ? 1 : 0;
      if (rc[2]) selp[I0] = 1.f;
    }
    __syncthreads();
    if (rc[2]) { mm[rc[0] * 32 + lane] = -10000.f; mm[lane * 32 + rc[1]] = -10000.f; }
    __syncthreads();
  }
}

__global__ void focus_k(const float* __restrict__ sim, const float* __restrict__ sel,
                        float* __restrict__ foc) {
  int idx = blockIdx.x * blockDim.x + threadIdx.x;
  if (idx >= Bz * 12 * 1024) return;
  int b = idx / (12 * 1024);
  int r = idx & 1023;
  float s9 = sel[((size_t)b * 2 + 0) * 1024 + r];
  float s10 = sel[((size_t)b * 2 + 1) * 1024 + r];
  float msk = (s9 + s10 > 0.f) ? 1.f : 0.1f;
  foc[idx] = sim[idx] * msk;
}

// ---------------------------------------------------------------------------
// Implicit-GEMM convolution with WMMA + LDS-staged im2col.
// One 16-row M tile per block; the block cooperatively builds the 16 x Kp
// im2col panel in LDS once, then each wave computes a 16x64 column group,
// reading A fragments from LDS (ds path) and f16 weights from global (L2-hot).
// B-weight loads are software-pipelined one K-stage ahead of the WMMA group.
// blockDim.x == 32 * (Np/64), so every wave owns a group (barrier-safe).
// ---------------------------------------------------------------------------
extern __shared__ _Float16 conv_panel[];

__global__ void conv_wmma(const float* __restrict__ in, const _Float16* __restrict__ Wt,
                          const float* __restrict__ bias, const float* __restrict__ res,
                          float* __restrict__ out, int Ci, int Hin, int Win,
                          int Co, int Np, int Ho, int Wo, int stride, int Kp, int relu) {
  int tid = threadIdx.x;
  int lane = tid & 31;
  int wv = tid >> 5;            // column-group index (Np/64 waves)
  int tm = blockIdx.x << 4;
  int hw = Ho * Wo;
  int Kreal = Ci * 9;
  // cooperative im2col staging of the 16 x Kp panel
  for (int i = tid; i < 16 * Kp; i += blockDim.x) {
    int r = i / Kp, k = i - r * Kp;
    int m = tm + r;
    int bA = m / hw;
    int r0 = m - bA * hw;
    int oy = r0 / Wo, ox = r0 - (r0 / Wo) * Wo;
    float v = 0.f;
    if (k < Kreal) {
      int ci = k / 9;
      int r9 = k - ci * 9;
      int ky = r9 / 3, kx = r9 - (r9 / 3) * 3;
      int iy = oy * stride + ky - 1, ix = ox * stride + kx - 1;
      if (iy >= 0 && iy < Hin && ix >= 0 && ix < Win)
        v = in[(((size_t)bA * Ci + ci) * Hin + iy) * Win + ix];
    }
    conv_panel[r * Kp + k] = (_Float16)v;
  }
  __syncthreads();

  int tn = wv << 6;
  v8f a0 = {}, a1 = {}, a2 = {}, a3 = {};
  v16h b0 = load_b_f16(Wt, Kp, tn,      lane, 0);
  v16h b1 = load_b_f16(Wt, Kp, tn + 16, lane, 0);
  v16h b2 = load_b_f16(Wt, Kp, tn + 32, lane, 0);
  v16h b3 = load_b_f16(Wt, Kp, tn + 48, lane, 0);
  int k0 = 0;
#pragma unroll 1
  for (; k0 + 32 < Kp; k0 += 32) {
    int kn = k0 + 32;
    v16h a   = load_a_f16(conv_panel, Kp, 0, lane, k0);   // LDS fragment (ds_load)
    v16h bn0 = load_b_f16(Wt, Kp, tn,      lane, kn);
    v16h bn1 = load_b_f16(Wt, Kp, tn + 16, lane, kn);
    v16h bn2 = load_b_f16(Wt, Kp, tn + 32, lane, kn);
    v16h bn3 = load_b_f16(Wt, Kp, tn + 48, lane, kn);
    a0 = WMMA_F16(a, b0, a0);
    a1 = WMMA_F16(a, b1, a1);
    a2 = WMMA_F16(a, b2, a2);
    a3 = WMMA_F16(a, b3, a3);
    b0 = bn0; b1 = bn1; b2 = bn2; b3 = bn3;
  }
  {
    v16h a = load_a_f16(conv_panel, Kp, 0, lane, k0);
    a0 = WMMA_F16(a, b0, a0);
    a1 = WMMA_F16(a, b1, a1);
    a2 = WMMA_F16(a, b2, a2);
    a3 = WMMA_F16(a, b3, a3);
  }
#pragma unroll
  for (int j = 0; j < 4; ++j) {
    v8f acc = (j == 0) ? a0 : (j == 1) ? a1 : (j == 2) ? a2 : a3;
    int n = tn + j * 16 + (lane & 15);
    if (n < Co) {
      float bvl = bias[n];
#pragma unroll
      for (int v = 0; v < 8; ++v) {
        int m2 = tm + v + ((lane >> 4) << 3);
        int b2 = m2 / hw;
        int r2 = m2 - b2 * hw;
        int y2 = r2 / Wo, x2 = r2 - (r2 / Wo) * Wo;
        size_t o = (((size_t)b2 * Co + n) * Ho + y2) * Wo + x2;
        float x = acc[v] + bvl;
        if (res) x += res[o];
        if (relu) x = fmaxf(x, 0.f);
        out[o] = x;
      }
    }
  }
}

__global__ void maxpool_k(const float* __restrict__ in, float* __restrict__ out, int C, int Hi) {
  int Ho = Hi >> 1;
  int n = Bz * C * Ho * Ho;
  int idx = blockIdx.x * blockDim.x + threadIdx.x;
  if (idx >= n) return;
  int x = idx % Ho;
  int t = idx / Ho;
  int y = t % Ho; t /= Ho;
  int c = t % C;
  int b = t / C;
  const float* p = in + (((size_t)b * C + c) * Hi + y * 2) * Hi + x * 2;
  out[idx] = fmaxf(fmaxf(p[0], p[1]), fmaxf(p[Hi], p[Hi + 1]));
}

// ---------------------------------------------------------------------------
// Weight / activation f16 conversion (with N->Np, K->Kp zero padding)
// ---------------------------------------------------------------------------
__global__ void cvt_w_k(const float* __restrict__ w, _Float16* __restrict__ o,
                        int N, int Kin, int Np, int Kp) {
  long idx = (long)blockIdx.x * blockDim.x + threadIdx.x;
  if (idx >= (long)Np * Kp) return;
  int n = (int)(idx / Kp), k = (int)(idx - (long)n * Kp);
  float v = (n < N && k < Kin) ? w[(size_t)n * Kin + k] : 0.f;
  o[idx] = (_Float16)v;
}

__global__ void cvt_f16_k(const float* __restrict__ in, _Float16* __restrict__ o, int n) {
  int idx = blockIdx.x * blockDim.x + threadIdx.x;
  if (idx < n) o[idx] = (_Float16)in[idx];
}

// Final (B,2) head: z = fc @ out_w^T + out_b; log_softmax.
__global__ void head_k(const float* __restrict__ fc, const float* __restrict__ ow,
                       const float* __restrict__ ob, float* __restrict__ out) {
  int b = blockIdx.x * blockDim.x + threadIdx.x;
  if (b >= Bz) return;
  float z0 = ob[0], z1 = ob[1];
  for (int k = 0; k < 128; ++k) {
    float x = fc[b * 128 + k];
    z0 += x * ow[k];
    z1 += x * ow[128 + k];
  }
  float mx = fmaxf(z0, z1);
  float l = logf(expf(z0 - mx) + expf(z1 - mx)) + mx;
  out[b * 2 + 0] = z0 - l;
  out[b * 2 + 1] = z1 - l;
}

// ---------------------------------------------------------------------------
// Host launch
// ---------------------------------------------------------------------------
struct Ptrs {
  const float *sent1, *sent2;
  const float *w1ih, *w1hh, *b1ih, *b1hh;
  const float *w2fih, *w2fhh, *b2fih, *b2fhh;
  const float *w2bih, *w2bhh, *b2bih, *b2bhh;
  const float *w3fih, *w3fhh, *b3fih, *b3fhh;
  const float *w3bih, *w3bhh, *b3bih, *b3bhh;
  const float *c1w, *c1b, *c2w, *c2b, *c3w, *c3b, *c4w, *c4b, *c5w, *c5b;
  const float *r1w, *r1b, *r2w, *r2b, *dw, *db, *ow, *ob;
};

static inline int cdiv(long a, long b) { return (int)((a + b - 1) / b); }
static inline int gblk64(int M, int N) { return (((M >> 4) * (N >> 6)) + 3) >> 2; }

extern "C" void kernel_launch(void* const* d_in, const int* in_sizes, int n_in,
                              void* d_out, int out_size, void* d_ws, size_t ws_size,
                              hipStream_t stream) {
  (void)n_in; (void)out_size; (void)ws_size;
  const float* const* F = (const float* const*)d_in;
  Ptrs P;
  if (in_sizes[0] == Bz * DIMv * Sq) {
    // insertion-order flattening
    P.sent1 = F[0];  P.sent2 = F[1];
    P.w1ih = F[2];  P.w1hh = F[3];  P.b1ih = F[4];  P.b1hh = F[5];
    P.w2fih = F[6]; P.w2fhh = F[7]; P.b2fih = F[8]; P.b2fhh = F[9];
    P.w2bih = F[10]; P.w2bhh = F[11]; P.b2bih = F[12]; P.b2bhh = F[13];
    P.w3fih = F[14]; P.w3fhh = F[15]; P.b3fih = F[16]; P.b3fhh = F[17];
    P.w3bih = F[18]; P.w3bhh = F[19]; P.b3bih = F[20]; P.b3bhh = F[21];
    P.c1w = F[22]; P.c1b = F[23]; P.c2w = F[24]; P.c2b = F[25];
    P.c3w = F[26]; P.c3b = F[27]; P.c4w = F[28]; P.c4b = F[29];
    P.c5w = F[30]; P.c5b = F[31];
    P.r1w = F[32]; P.r1b = F[33]; P.r2w = F[34]; P.r2b = F[35];
    P.dw = F[36]; P.db = F[37]; P.ow = F[38]; P.ob = F[39];
  } else {
    // sorted pytree flattening (params first, keys sorted)
    P.c1b = F[0];  P.c1w = F[1];  P.c2b = F[2];  P.c2w = F[3];
    P.c3b = F[4];  P.c3w = F[5];  P.c4b = F[6];  P.c4w = F[7];
    P.c5b = F[8];  P.c5w = F[9];  P.db = F[10];  P.dw = F[11];
    P.ob = F[12];  P.ow = F[13];  P.r1b = F[14]; P.r1w = F[15];
    P.r2b = F[16]; P.r2w = F[17];
    P.w1hh = F[18]; P.w1ih = F[19]; P.b1hh = F[20]; P.b1ih = F[21];
    P.w2bhh = F[22]; P.w2bih = F[23]; P.b2bhh = F[24]; P.b2bih = F[25];
    P.w2fhh = F[26]; P.w2fih = F[27]; P.b2fhh = F[28]; P.b2fih = F[29];
    P.w3bhh = F[30]; P.w3bih = F[31]; P.b3bhh = F[32]; P.b3bih = F[33];
    P.w3fhh = F[34]; P.w3fih = F[35]; P.b3fhh = F[36]; P.b3fih = F[37];
    P.sent1 = F[38]; P.sent2 = F[39];
  }

  char* W = (char*)d_ws;
  auto H16 = [&](size_t o) { return (_Float16*)(W + o); };
  auto F32 = [&](size_t o) { return (float*)(W + o); };

  // ---- weight conversion to f16 (padded) -------------------------------
  auto cvtw = [&](const float* src, size_t dst, int N, int Kin, int Np, int Kp) {
    cvt_w_k<<<cdiv((long)Np * Kp, 256), 256, 0, stream>>>(src, H16(dst), N, Kin, Np, Kp);
  };
  cvtw(P.w1ih, O_W1IH, 1024, 300, 1024, 320);
  cvtw(P.w1hh, O_W1HH, 1024, 256, 1024, 256);
  cvtw(P.w2fih, O_W2FIH, 1024, 812, 1024, 832);
  cvtw(P.w2bih, O_W2BIH, 1024, 812, 1024, 832);
  cvtw(P.w2fhh, O_W2FHH, 1024, 256, 1024, 256);
  cvtw(P.w2bhh, O_W2BHH, 1024, 256, 1024, 256);
  cvtw(P.w3fih, O_W3FIH, 1024, 1324, 1024, 1344);
  cvtw(P.w3bih, O_W3BIH, 1024, 1324, 1024, 1344);
  cvtw(P.w3fhh, O_W3FHH, 1024, 256, 1024, 256);
  cvtw(P.w3bhh, O_W3BHH, 1024, 256, 1024, 256);
  cvtw(P.c1w, O_CW1, 128, 108, 128, 128);
  cvtw(P.c2w, O_CW2, 164, 1152, 192, 1152);
  cvtw(P.c3w, O_CW3, 192, 1476, 192, 1504);
  cvtw(P.c4w, O_CW4, 192, 1728, 192, 1728);
  cvtw(P.c5w, O_CW5, 128, 1728, 128, 1728);
  cvtw(P.r1w, O_RW1, 164, 108, 192, 128);
  cvtw(P.r2w, O_RW2, 192, 1476, 192, 1504);
  cvtw(P.dw, O_DW, 128, 128, 128, 128);

  // ---- layer 1 LSTM (batch 512: x1, x2, flip(x1), flip(x2)) ------------
  asm_a1_k<<<cdiv((long)M1 * K1P, 256), 256, 0, stream>>>(P.sent1, P.sent2, H16(O_A1));
  gemm_f16<<<gblk64(M1, G4), 128, 0, stream>>>(H16(O_A1), H16(O_W1IH), F32(O_GIN),
                                               P.b1ih, P.b1hh, M1, G4, K1P, 0);
  for (int t = 0; t < Sq; ++t) {
    lstm_step_gemm<<<gblk64(512, G4), 128, 0, stream>>>(F32(O_GIN), H16(O_H1), H16(O_W1HH),
                                                        F32(O_GT), 512, t);
    lstm_pointwise<<<cdiv(512 * Hh, 256), 256, 0, stream>>>(F32(O_GT), F32(O_CS), H16(O_H1), 512, t);
  }

  // ---- layer 2 bi-LSTM (batch 256 each direction) ----------------------
  asm_a2_k<<<cdiv((long)M2 * K2P, 256), 256, 0, stream>>>(P.sent1, P.sent2, H16(O_H1), H16(O_A2F), 0);
  asm_a2_k<<<cdiv((long)M2 * K2P, 256), 256, 0, stream>>>(P.sent1, P.sent2, H16(O_H1), H16(O_A2B), 1);
  gemm_f16<<<gblk64(M2, G4), 128, 0, stream>>>(H16(O_A2F), H16(O_W2FIH), F32(O_GIN2F),
                                               P.b2fih, P.b2fhh, M2, G4, K2P, 0);
  gemm_f16<<<gblk64(M2, G4), 128, 0, stream>>>(H16(O_A2B), H16(O_W2BIH), F32(O_GIN2B),
                                               P.b2bih, P.b2bhh, M2, G4, K2P, 0);
  for (int t = 0; t < Sq; ++t) {
    lstm_step_gemm<<<gblk64(256, G4), 128, 0, stream>>>(F32(O_GIN2F), H16(O_H2F), H16(O_W2FHH),
                                                        F32(O_GT), 256, t);
    lstm_pointwise<<<cdiv(256 * Hh, 256), 256, 0, stream>>>(F32(O_GT), F32(O_CS), H16(O_H2F), 256, t);
  }
  for (int t = 0; t < Sq; ++t) {
    lstm_step_gemm<<<gblk64(256, G4), 128, 0, stream>>>(F32(O_GIN2B), H16(O_H2B), H16(O_W2BHH),
                                                        F32(O_GT), 256, t);
    lstm_pointwise<<<cdiv(256 * Hh, 256), 256, 0, stream>>>(F32(O_GT), F32(O_CS), H16(O_H2B), 256, t);
  }

  // ---- layer 3 bi-LSTM -------------------------------------------------
  asm_a3_k<<<cdiv((long)M2 * K3P, 256), 256, 0, stream>>>(P.sent1, P.sent2, H16(O_H1),
                                                          H16(O_H2F), H16(O_H2B), H16(O_A3F), 0);
  asm_a3_k<<<cdiv((long)M2 * K3P, 256), 256, 0, stream>>>(P.sent1, P.sent2, H16(O_H1),
                                                          H16(O_H2F), H16(O_H2B), H16(O_A3B), 1);
  gemm_f16<<<gblk64(M2, G4), 128, 0, stream>>>(H16(O_A3F), H16(O_W3FIH), F32(O_GIN2F),
                                               P.b3fih, P.b3fhh, M2, G4, K3P, 0);
  gemm_f16<<<gblk64(M2, G4), 128, 0, stream>>>(H16(O_A3B), H16(O_W3BIH), F32(O_GIN2B),
                                               P.b3bih, P.b3bhh, M2, G4, K3P, 0);
  for (int t = 0; t < Sq; ++t) {
    lstm_step_gemm<<<gblk64(256, G4), 128, 0, stream>>>(F32(O_GIN2F), H16(O_H3F), H16(O_W3FHH),
                                                        F32(O_GT), 256, t);
    lstm_pointwise<<<cdiv(256 * Hh, 256), 256, 0, stream>>>(F32(O_GT), F32(O_CS), H16(O_H3F), 256, t);
  }
  for (int t = 0; t < Sq; ++t) {
    lstm_step_gemm<<<gblk64(256, G4), 128, 0, stream>>>(F32(O_GIN2B), H16(O_H3B), H16(O_W3BHH),
                                                        F32(O_GT), 256, t);
    lstm_pointwise<<<cdiv(256 * Hh, 256), 256, 0, stream>>>(F32(O_GT), F32(O_CS), H16(O_H3B), 256, t);
  }

  // ---- similarity cube -------------------------------------------------
  build_seq_k<<<cdiv(Bz * Sq * 512, 256), 256, 0, stream>>>(H16(O_H3F), H16(O_H3B), H16(O_SEQ1), 0);
  build_seq_k<<<cdiv(Bz * Sq * 512, 256), 256, 0, stream>>>(H16(O_H3F), H16(O_H3B), H16(O_SEQ2), 1);
  dots_wmma_k<<<cdiv(Bz * 16, 4), 128, 0, stream>>>(H16(O_SEQ1), H16(O_SEQ2), F32(O_DOTS));
  rowstats_k<<<cdiv(2 * Bz * 32, 256), 256, 0, stream>>>(H16(O_SEQ1), H16(O_SEQ2), F32(O_ST));
  sim_assemble_k<<<cdiv(Bz * 1024, 256), 256, 0, stream>>>(F32(O_DOTS), F32(O_ST), F32(O_SIM));
  greedy_k<<<Bz * 2, 32, 0, stream>>>(F32(O_SIM), F32(O_SEL));
  focus_k<<<cdiv(Bz * 12 * 1024, 256), 256, 0, stream>>>(F32(O_SIM), F32(O_SEL), F32(O_FOC));

  // ---- CNN classifier (LDS-staged implicit GEMM) -----------------------
  // conv1: (B,12,32,32)->(B,128,32,32), relu, pool -> (B,128,16,16)
  conv_wmma<<<Bz * 32 * 32 / 16, 64, 16 * 128 * 2, stream>>>(F32(O_FOC), H16(O_CW1), P.c1b,
      nullptr, F32(O_C1O), 12, 32, 32, 128, 128, 32, 32, 1, 128, 1);
  maxpool_k<<<cdiv(Bz * 128 * 16 * 16, 256), 256, 0, stream>>>(F32(O_C1O), F32(O_P1), 128, 32);
  // resconv1: (B,12,32,32) stride2 -> (B,164,16,16)
  conv_wmma<<<Bz * 16 * 16 / 16, 96, 16 * 128 * 2, stream>>>(F32(O_FOC), H16(O_RW1), P.r1b,
      nullptr, F32(O_R1O), 12, 32, 32, 164, 192, 16, 16, 2, 128, 0);
  // conv2 + residual + relu, pool -> (B,164,8,8)
  conv_wmma<<<Bz * 16 * 16 / 16, 96, 16 * 1152 * 2, stream>>>(F32(O_P1), H16(O_CW2), P.c2b,
      F32(O_R1O), F32(O_C2O), 128, 16, 16, 164, 192, 16, 16, 1, 1152, 1);
  maxpool_k<<<cdiv(Bz * 164 * 8 * 8, 256), 256, 0, stream>>>(F32(O_C2O), F32(O_P2), 164, 16);
  // conv3, relu, pool -> (B,192,4,4)
  conv_wmma<<<Bz * 8 * 8 / 16, 96, 16 * 1504 * 2, stream>>>(F32(O_P2), H16(O_CW3), P.c3b,
      nullptr, F32(O_C3O), 164, 8, 8, 192, 192, 8, 8, 1, 1504, 1);
  maxpool_k<<<cdiv(Bz * 192 * 4 * 4, 256), 256, 0, stream>>>(F32(O_C3O), F32(O_P3), 192, 8);
  // resconv2: (B,164,8,8) stride2 -> (B,192,4,4)
  conv_wmma<<<Bz * 4 * 4 / 16, 96, 16 * 1504 * 2, stream>>>(F32(O_P2), H16(O_RW2), P.r2b,
      nullptr, F32(O_R2O), 164, 8, 8, 192, 192, 4, 4, 2, 1504, 0);
  // conv4 + residual + relu, pool -> (B,192,2,2)
  conv_wmma<<<Bz * 4 * 4 / 16, 96, 16 * 1728 * 2, stream>>>(F32(O_P3), H16(O_CW4), P.c4b,
      F32(O_R2O), F32(O_C4O), 192, 4, 4, 192, 192, 4, 4, 1, 1728, 1);
  maxpool_k<<<cdiv(Bz * 192 * 2 * 2, 256), 256, 0, stream>>>(F32(O_C4O), F32(O_P4), 192, 4);
  // conv5, relu, pool -> (B,128,1,1)
  conv_wmma<<<Bz * 2 * 2 / 16, 64, 16 * 1728 * 2, stream>>>(F32(O_P4), H16(O_CW5), P.c5b,
      nullptr, F32(O_C5O), 192, 2, 2, 128, 128, 2, 2, 1, 1728, 1);
  maxpool_k<<<cdiv(Bz * 128, 256), 256, 0, stream>>>(F32(O_C5O), F32(O_P5), 128, 2);

  // ---- head ------------------------------------------------------------
  cvt_f16_k<<<cdiv(Bz * 128, 256), 256, 0, stream>>>(F32(O_P5), H16(O_FH), Bz * 128);
  gemm_f16<<<gblk64(Bz, 128), 128, 0, stream>>>(H16(O_FH), H16(O_DW), F32(O_FC),
                                                P.db, nullptr, Bz, 128, 128, 1);
  head_k<<<1, 128, 0, stream>>>(F32(O_FC), P.ow, P.ob, (float*)d_out);
}
